// Attention_11106785428044
// MI455X (gfx1250) — matrix operand order repaired and checked by reading, not compiled, not run on hardware
//
#include <hip/hip_runtime.h>

typedef unsigned short u16;
typedef unsigned int u32;
typedef __attribute__((ext_vector_type(16))) __bf16 v16bf;
typedef __attribute__((ext_vector_type(8)))  float  v8f;

#define H_  16
#define DH_ 64
#define B_  4
#define N_  1024
#define DIM_ 1024
#define ROWS_ (B_ * N_)   // 4096

// ---------- CDNA5 data-mover feature detection ----------
#if __has_builtin(__builtin_amdgcn_tensor_load_to_lds)
#define HAS_TDM 1
#else
#define HAS_TDM 0
#endif
#if __has_builtin(__builtin_amdgcn_global_load_async_to_lds_b128)
#define HAS_ASYNC 1
#else
#define HAS_ASYNC 0
#endif

#if HAS_TDM
typedef __attribute__((ext_vector_type(4))) unsigned int u32x4;
typedef __attribute__((ext_vector_type(4))) int i32x4;
typedef __attribute__((ext_vector_type(8))) int i32x8;

// 2D tile load via Tensor Data Mover (6-arg builtin: g0, g1, g2, g3, g4, cpol).
//   tile0       : elements per row (bf16)
//   tile1       : number of rows
//   strideElems : tensor row stride (elements) == tensor_dim0 bound
//   tensorD1    : tensor rows (OOB bound)
//   padI/padA   : LDS pad interval/amount encodings (ISA D# group1)
__device__ __forceinline__ void tdm_load_2d(u16* lds, const u16* gsrc,
                                            u32 tile0, u32 tile1, u32 strideElems,
                                            u32 tensorD1, u32 padI, u32 padA) {
    unsigned long long ga = (unsigned long long)gsrc;
    u32x4 g0;
    g0[0] = 1u;                                   // count=1, user descriptor
    g0[1] = (u32)(unsigned long long)lds;         // LDS byte address (flat[31:0])
    g0[2] = (u32)ga;                              // global_addr[31:0]
    g0[3] = (u32)((ga >> 32) & 0x1FFFFFFull) | (2u << 30);  // addr[56:32] | type=2
    u32 td0 = strideElems;
    i32x8 g1;
    g1[0] = (int)((1u << 16) | (1u << 20) | (padI << 22) | (padA << 25)); // 2B elems, pad_enable
    g1[1] = (int)((td0 & 0xffffu) << 16);                    // tensor_dim0 lo16
    g1[2] = (int)((td0 >> 16) | ((tensorD1 & 0xffffu) << 16));
    g1[3] = (int)((tensorD1 >> 16) | (tile0 << 16));         // tile_dim0
    g1[4] = (int)tile1;                                      // tile_dim1 (tile_dim2=0)
    g1[5] = (int)strideElems;                                // tensor_dim0_stride lo32
    g1[6] = 0;
    g1[7] = 0;
    i32x4 z4 = {0, 0, 0, 0};
    i32x8 z8 = {0, 0, 0, 0, 0, 0, 0, 0};
    __builtin_amdgcn_tensor_load_to_lds(g0, g1, z4, z4, z8, 0);
}
__device__ __forceinline__ void tdm_wait() {
    __builtin_amdgcn_s_wait_tensorcnt(0);
}
#endif

// 16-byte global -> LDS copy (async if available)
__device__ __forceinline__ void cp16_g2l(u16* l, const u16* g) {
#if HAS_ASYNC && !HAS_TDM
    typedef __attribute__((address_space(1))) void as1_void;
    typedef __attribute__((address_space(3))) void as3_void;
    __builtin_amdgcn_global_load_async_to_lds_b128(
        (as1_void*)(unsigned long long)g,
        (as3_void*)(u32)(unsigned long long)l, 0, 0);
#else
    *(uint4*)l = *(const uint4*)g;
#endif
}
__device__ __forceinline__ void async_wait() {
#if HAS_ASYNC && !HAS_TDM
#if __has_builtin(__builtin_amdgcn_s_wait_asynccnt)
    __builtin_amdgcn_s_wait_asynccnt(0);
#else
    asm volatile("s_wait_asynccnt 0" ::: "memory");
#endif
#endif
}

// ---------- bf16 helpers (raw u16 storage) ----------
__device__ __forceinline__ u16 f2bf(float f) {
    union { float f; u32 u; } c; c.f = f;
    u32 u = c.u;
    u += 0x7fffu + ((u >> 16) & 1u);   // round-to-nearest-even
    return (u16)(u >> 16);
}
__device__ __forceinline__ float bf2f(u16 h) {
    union { u32 u; float f; } c; c.u = ((u32)h) << 16; return c.f;
}
__device__ __forceinline__ v16bf make_frag(const u16* p0, const u16* p1) {
    union { struct { uint4 a, b; } s; v16bf v; } u;
    u.s.a = *(const uint4*)p0;
    u.s.b = *(const uint4*)p1;
    return u.v;
}
__device__ __forceinline__ v8f wmma_bf16(v16bf a, v16bf b, v8f c) {
    return __builtin_amdgcn_wmma_f32_16x16x32_bf16(false, a, false, b, (short)0, c, false, false);
}

// ---------- kernel 0: transpose + cast fp32 -> bf16 : out[c*R + r] = in[r*C + c] ----------
__global__ __launch_bounds__(256) void k_transpose_cast(const float* __restrict__ in,
                                                        u16* __restrict__ out, int R, int C) {
    int idx = blockIdx.x * 256 + threadIdx.x;
    if (idx >= R * C) return;
    int r = idx / C, c = idx % C;
    out[(size_t)c * R + r] = f2bf(in[idx]);
}

// ---------- kernel 1: layernorm + cast, one wave per row ----------
__global__ __launch_bounds__(256) void k_ln_cast(const float* __restrict__ x,
                                                 const float* __restrict__ w,
                                                 const float* __restrict__ b,
                                                 u16* __restrict__ y, int rows, int D) {
    int wave = threadIdx.x >> 5, lane = threadIdx.x & 31;
    int row = blockIdx.x * 8 + wave;
    if (row >= rows) return;
    const float* xr = x + (size_t)row * D;
    float s = 0.f, s2 = 0.f;
    for (int i = lane; i < D; i += 32) { float v = xr[i]; s += v; s2 += v * v; }
    for (int o = 16; o > 0; o >>= 1) { s += __shfl_xor(s, o, 32); s2 += __shfl_xor(s2, o, 32); }
    float mu = s / D;
    float var = s2 / D - mu * mu;
    float inv = rsqrtf(var + 1e-5f);
    u16* yr = y + (size_t)row * D;
    for (int i = lane; i < D; i += 32)
        yr[i] = f2bf((xr[i] - mu) * inv * w[i] + b[i]);
}

// ---------- GEMM tile staging: 128 rows x 32 bf16, LDS stride 40 ----------
__device__ __forceinline__ void gemm_stage(u16* sA, u16* sB,
                                           const u16* __restrict__ A, const u16* __restrict__ Bt,
                                           int bm0, int bn0, int K, int k0, int tid, int wave) {
#if HAS_TDM
    if (wave == 0) {
        // rows of 32 elems = 16 dwords, pad +4 dwords -> LDS stride 40 elems
        tdm_load_2d(sA, A + (size_t)bm0 * K + k0, 32, 128, (u32)K, (u32)ROWS_, 3, 3);
        tdm_load_2d(sB, Bt + (size_t)bn0 * K + k0, 32, 128, (u32)K, (u32)DIM_, 3, 3);
    }
#else
    for (int i = 0; i < 2; ++i) {
        int g = tid * 2 + i;
        int r = g >> 2, kq = (g & 3) * 8;
        cp16_g2l(&sA[r * 40 + kq], &A[(size_t)(bm0 + r) * K + k0 + kq]);
        cp16_g2l(&sB[r * 40 + kq], &Bt[(size_t)(bn0 + r) * K + k0 + kq]);
    }
#endif
}
__device__ __forceinline__ void stage_wait(int wave) {
#if HAS_TDM
    if (wave == 0) tdm_wait();
#else
    async_wait();
#endif
}

// ---------- kernel 2: bf16 GEMM  C[M,Nc] = A[M,K] @ Bt[Nc,K] (+bias) ----------
// 128x128 block tile, 256 threads (8 waves in 4x2), BK=32, double-buffered LDS
template <bool OUT_F32, bool BIAS>
__global__ __launch_bounds__(256) void k_gemm_bf16(const u16* __restrict__ A,
                                                   const u16* __restrict__ Bt,
                                                   const float* __restrict__ bias,
                                                   void* __restrict__ Cout,
                                                   int M, int Nc, int K) {
    __shared__ __align__(16) u16 sA[2][128 * 40];
    __shared__ __align__(16) u16 sB[2][128 * 40];
    const int tid = threadIdx.x, wave = tid >> 5, lane = tid & 31;
    const int l15 = lane & 15, lh = lane >> 4;
    const int wm = wave >> 1, wn = wave & 1;
    const int bn0 = blockIdx.x * 128, bm0 = blockIdx.y * 128;

    v8f acc[8] = {};
    const int nk = K / 32;
    gemm_stage(sA[0], sB[0], A, Bt, bm0, bn0, K, 0, tid, wave);
    int cur = 0;
    for (int kt = 0; kt < nk; ++kt) {
        stage_wait(wave);
        __syncthreads();
        if (kt + 1 < nk)
            gemm_stage(sA[cur ^ 1], sB[cur ^ 1], A, Bt, bm0, bn0, K, (kt + 1) * 32, tid, wave);
        const u16* cA = sA[cur];
        const u16* cB = sB[cur];
        v16bf bfrag[4];
        for (int tn = 0; tn < 4; ++tn) {
            int col = wn * 64 + tn * 16 + l15;
            bfrag[tn] = make_frag(&cB[col * 40 + lh * 16], &cB[col * 40 + lh * 16 + 8]);
        }
        for (int tm = 0; tm < 2; ++tm) {
            int row = wm * 32 + tm * 16 + l15;
            v16bf af = make_frag(&cA[row * 40 + lh * 16], &cA[row * 40 + lh * 16 + 8]);
            for (int tn = 0; tn < 4; ++tn)
                acc[tm * 4 + tn] = wmma_bf16(af, bfrag[tn], acc[tm * 4 + tn]);
        }
        cur ^= 1;
    }
    for (int tm = 0; tm < 2; ++tm)
        for (int tn = 0; tn < 4; ++tn) {
            v8f c = acc[tm * 4 + tn];
            int colg = bn0 + wn * 64 + tn * 16 + l15;
            float bv = BIAS ? bias[colg] : 0.f;
            for (int r = 0; r < 8; ++r) {
                int rowg = bm0 + wm * 32 + tm * 16 + r + lh * 8;
                float val = c[r] + bv;
                if (OUT_F32) ((float*)Cout)[(size_t)rowg * Nc + colg] = val;
                else         ((u16*)Cout)[(size_t)rowg * Nc + colg] = f2bf(val);
            }
        }
}

// ---------- kernel 3: zero fp32 ----------
__global__ __launch_bounds__(256) void k_zero_f32(float* __restrict__ p, int n) {
    int i = blockIdx.x * 256 + threadIdx.x;
    if (i < n) p[i] = 0.f;
}

// ---------- kernel 4: per (h,b,n) stats ----------
__global__ __launch_bounds__(32) void k_stats(const u16* __restrict__ f,
                                              float* __restrict__ mean_out,
                                              float* __restrict__ invn_out,
                                              float* __restrict__ pooled, int poolOff) {
    int h = blockIdx.x, bn = blockIdx.y, lane = threadIdx.x;
    int b = bn / N_, n = bn % N_;
    const u16* p = f + (size_t)bn * DIM_ + h * DH_ + lane * 2;
    float v0 = bf2f(p[0]), v1 = bf2f(p[1]);
    float s = v0 + v1, s2 = v0 * v0 + v1 * v1;
    for (int o = 16; o > 0; o >>= 1) { s += __shfl_xor(s, o, 32); s2 += __shfl_xor(s2, o, 32); }
    int hb = h * B_ + b;
    if (lane == 0) {
        mean_out[(size_t)hb * N_ + n] = s * (1.f / DH_);
        invn_out[(size_t)hb * N_ + n] = rsqrtf(s2 + 1e-20f);
    }
    float* pb = pooled + (size_t)hb * (2 * DH_) + poolOff + lane * 2;
    atomicAdd(&pb[0], v0 * (1.f / N_));
    atomicAdd(&pb[1], v1 * (1.f / N_));
}

// ---------- kernel 5: gate MLP per (h,b) ----------
__global__ __launch_bounds__(64) void k_gate(const float* __restrict__ pooled,
                                             const float* __restrict__ w1, const float* __restrict__ b1,
                                             const float* __restrict__ lnw, const float* __restrict__ lnb,
                                             const float* __restrict__ w2, const float* __restrict__ b2,
                                             float* __restrict__ wgate) {
    int hb = blockIdx.x, d = threadIdx.x;
    __shared__ float feat[2 * DH_];
    __shared__ float red[DH_];
    feat[d]       = pooled[(size_t)hb * (2 * DH_) + d];
    feat[d + DH_] = pooled[(size_t)hb * (2 * DH_) + DH_ + d];
    __syncthreads();
    float acc = b1[d];
    for (int j = 0; j < 2 * DH_; ++j) acc += feat[j] * w1[j * DH_ + d];
    red[d] = acc; __syncthreads();
    for (int o = 32; o > 0; o >>= 1) { if (d < o) red[d] += red[d + o]; __syncthreads(); }
    float mu = red[0] * (1.f / DH_); __syncthreads();
    red[d] = (acc - mu) * (acc - mu); __syncthreads();
    for (int o = 32; o > 0; o >>= 1) { if (d < o) red[d] += red[d + o]; __syncthreads(); }
    float var = red[0] * (1.f / DH_); __syncthreads();
    float lnv = (acc - mu) * rsqrtf(var + 1e-5f) * lnw[d] + lnb[d];
    red[d] = fmaxf(lnv, 0.f) * w2[d]; __syncthreads();
    for (int o = 32; o > 0; o >>= 1) { if (d < o) red[d] += red[d + o]; __syncthreads(); }
    if (d == 0) {
        float g = red[0] + b2[0];
        wgate[hb] = 1.f / (1.f + __expf(-g));
    }
}

// ---------- kernel 6: fused attention per (h,b) ----------
// dots = S * [(1-w)*iqn*ikn + w/DH] - w*qm*km ;  O = dots @ V  (flash-style, KT=64)
__global__ __launch_bounds__(256) void k_attn(const u16* __restrict__ fq,
                                              const u16* __restrict__ fk,
                                              const u16* __restrict__ fv,
                                              const float* __restrict__ qm_g,
                                              const float* __restrict__ iqn_g,
                                              const float* __restrict__ km_g,
                                              const float* __restrict__ ikn_g,
                                              const float* __restrict__ wgate,
                                              u16* __restrict__ O) {
    __shared__ __align__(16) u16 sQ[128 * 72];
    __shared__ __align__(16) u16 sK[64 * 72];
    __shared__ __align__(16) u16 sVt[64 * 72];   // transposed: [d][k]
    __shared__ __align__(16) u16 sS[128 * 72];
    __shared__ float s_iqn[128], s_qm[128], s_ikn[64], s_km[64];

    const int hb = blockIdx.y, h = hb / B_, b = hb % B_;
    const int n0 = blockIdx.x * 128;
    const int tid = threadIdx.x, wave = tid >> 5, lane = tid & 31;
    const int l15 = lane & 15, lh = lane >> 4;
    const int wm = wave >> 1, wn = wave & 1;

    const float w = wgate[hb];
    const float cw = 1.f - w, wdh = w * (1.f / DH_);

    // stage Q tile: 128 rows x 64 bf16, LDS stride 72 (32dw rows + 4dw pad)
#if HAS_TDM
    if (wave == 0)
        tdm_load_2d(sQ, fq + (size_t)(b * N_ + n0) * DIM_ + h * DH_, 64, 128,
                    (u32)DIM_, (u32)ROWS_, 4, 3);
#else
    for (int i = 0; i < 4; ++i) {
        int g = tid * 4 + i;
        int r = g >> 3, dq = (g & 7) * 8;
        cp16_g2l(&sQ[r * 72 + dq],
                 &fq[((size_t)(b * N_ + n0 + r)) * DIM_ + h * DH_ + dq]);
    }
#endif
    if (tid < 128) {
        s_iqn[tid] = iqn_g[(size_t)hb * N_ + n0 + tid];
        s_qm[tid]  = qm_g[(size_t)hb * N_ + n0 + tid];
    }

    v8f oacc[4] = {};
    for (int m0 = 0; m0 < N_; m0 += 64) {
        __syncthreads();   // previous iteration done with sK/sVt/sS
#if HAS_TDM
        if (wave == 0)
            tdm_load_2d(sK, fk + (size_t)(b * N_ + m0) * DIM_ + h * DH_, 64, 64,
                        (u32)DIM_, (u32)ROWS_, 4, 3);
        for (int i = 0; i < 2; ++i) {   // V transposed scatter (TDM cannot transpose)
            int g = tid * 2 + i;
            int r = g >> 3, dq = (g & 7) * 8;
            uint4 vv = *(const uint4*)&fv[((size_t)(b * N_ + m0 + r)) * DIM_ + h * DH_ + dq];
            const u16* pv = (const u16*)&vv;
            for (int j = 0; j < 8; ++j) sVt[(dq + j) * 72 + r] = pv[j];
        }
#else
        for (int i = 0; i < 2; ++i) {
            int g = tid * 2 + i;
            int r = g >> 3, dq = (g & 7) * 8;
            cp16_g2l(&sK[r * 72 + dq],
                     &fk[((size_t)(b * N_ + m0 + r)) * DIM_ + h * DH_ + dq]);
            uint4 vv = *(const uint4*)&fv[((size_t)(b * N_ + m0 + r)) * DIM_ + h * DH_ + dq];
            const u16* pv = (const u16*)&vv;
            for (int j = 0; j < 8; ++j) sVt[(dq + j) * 72 + r] = pv[j];
        }
#endif
        if (tid < 64) {
            s_ikn[tid] = ikn_g[(size_t)hb * N_ + m0 + tid];
            s_km[tid]  = km_g[(size_t)hb * N_ + m0 + tid];
        }
        stage_wait(wave);
        __syncthreads();

        // S = Q @ K^T : wave tile 32x32 of the 128x64 S-tile
        v8f sacc[4] = {};
        for (int kk = 0; kk < DH_; kk += 32) {
            v16bf bf_[2];
            for (int tn = 0; tn < 2; ++tn) {
                int m = wn * 32 + tn * 16 + l15;
                bf_[tn] = make_frag(&sK[m * 72 + kk + lh * 16], &sK[m * 72 + kk + lh * 16 + 8]);
            }
            for (int tm = 0; tm < 2; ++tm) {
                int qr = wm * 32 + tm * 16 + l15;
                v16bf af = make_frag(&sQ[qr * 72 + kk + lh * 16], &sQ[qr * 72 + kk + lh * 16 + 8]);
                for (int tn = 0; tn < 2; ++tn)
                    sacc[tm * 2 + tn] = wmma_bf16(af, bf_[tn], sacc[tm * 2 + tn]);
            }
        }
        // elementwise transform -> bf16 S in LDS
        for (int tm = 0; tm < 2; ++tm)
            for (int tn = 0; tn < 2; ++tn) {
                v8f c = sacc[tm * 2 + tn];
                int nn = wn * 32 + tn * 16 + l15;
                float ik = s_ikn[nn], km = s_km[nn];
                for (int r = 0; r < 8; ++r) {
                    int mm = wm * 32 + tm * 16 + r + lh * 8;
                    float scale = cw * s_iqn[mm] * ik + wdh;
                    float val = c[r] * scale - w * s_qm[mm] * km;
                    sS[mm * 72 + nn] = f2bf(val);
                }
            }
        __syncthreads();

        // O += S @ V : wave tile 32x32 of the 128x64 O-tile, K = 64 keys
        for (int kk = 0; kk < 64; kk += 32) {
            v16bf bv[2];
            for (int tn = 0; tn < 2; ++tn) {
                int d = wn * 32 + tn * 16 + l15;
                bv[tn] = make_frag(&sVt[d * 72 + kk + lh * 16], &sVt[d * 72 + kk + lh * 16 + 8]);
            }
            for (int tm = 0; tm < 2; ++tm) {
                int orow = wm * 32 + tm * 16 + l15;
                v16bf af = make_frag(&sS[orow * 72 + kk + lh * 16], &sS[orow * 72 + kk + lh * 16 + 8]);
                for (int tn = 0; tn < 2; ++tn)
                    oacc[tm * 2 + tn] = wmma_bf16(af, bv[tn], oacc[tm * 2 + tn]);
            }
        }
    }
    // write O (bf16) to [(b,n), h*64+d]
    for (int tm = 0; tm < 2; ++tm)
        for (int tn = 0; tn < 2; ++tn) {
            v8f c = oacc[tm * 2 + tn];
            int d = wn * 32 + tn * 16 + l15;
            for (int r = 0; r < 8; ++r) {
                int mm = wm * 32 + tm * 16 + r + lh * 8;
                O[((size_t)(b * N_ + n0 + mm)) * DIM_ + h * DH_ + d] = f2bf(c[r]);
            }
        }
}

// ---------- workspace layout (bytes) ----------
#define OFF_LNQ   (size_t)0              // 8 MB  (reused as O after projections)
#define OFF_LNK   (size_t)(8u  << 20)
#define OFF_LNV   (size_t)(16u << 20)
#define OFF_FQ    (size_t)(24u << 20)
#define OFF_FK    (size_t)(32u << 20)
#define OFF_FV    (size_t)(40u << 20)
#define OFF_WINT  (size_t)(48u << 20)    // 2 MB
#define OFF_WOUTT (size_t)(50u << 20)    // 2 MB
#define OFF_QM    (size_t)(52u << 20)    // 256 KB each
#define OFF_IQN   (OFF_QM  + (256u << 10))
#define OFF_KM    (OFF_IQN + (256u << 10))
#define OFF_IKN   (OFF_KM  + (256u << 10))
#define OFF_POOL  (OFF_IKN + (256u << 10))   // 32 KB
#define OFF_WG    (OFF_POOL + (32u << 10))   // 256 B

extern "C" void kernel_launch(void* const* d_in, const int* in_sizes, int n_in,
                              void* d_out, int out_size, void* d_ws, size_t ws_size,
                              hipStream_t stream) {
    const float* q     = (const float*)d_in[0];
    const float* k     = (const float*)d_in[1];
    const float* v     = (const float*)d_in[2];
    const float* ln_w  = (const float*)d_in[3];
    const float* ln_b  = (const float*)d_in[4];
    const float* w_in  = (const float*)d_in[5];
    const float* wp_w1 = (const float*)d_in[6];
    const float* wp_b1 = (const float*)d_in[7];
    const float* wp_lw = (const float*)d_in[8];
    const float* wp_lb = (const float*)d_in[9];
    const float* wp_w2 = (const float*)d_in[10];
    const float* wp_b2 = (const float*)d_in[11];
    const float* w_out = (const float*)d_in[12];
    const float* b_out = (const float*)d_in[13];

    char* ws = (char*)d_ws;
    u16*   lnq   = (u16*)(ws + OFF_LNQ);
    u16*   lnk   = (u16*)(ws + OFF_LNK);
    u16*   lnv   = (u16*)(ws + OFF_LNV);
    u16*   fqb   = (u16*)(ws + OFF_FQ);
    u16*   fkb   = (u16*)(ws + OFF_FK);
    u16*   fvb   = (u16*)(ws + OFF_FV);
    u16*   winT  = (u16*)(ws + OFF_WINT);
    u16*   woutT = (u16*)(ws + OFF_WOUTT);
    float* qm    = (float*)(ws + OFF_QM);
    float* iqn   = (float*)(ws + OFF_IQN);
    float* km    = (float*)(ws + OFF_KM);
    float* ikn   = (float*)(ws + OFF_IKN);
    float* pool  = (float*)(ws + OFF_POOL);
    float* wgate = (float*)(ws + OFF_WG);
    u16*   Obuf  = lnq;   // reuse: LN buffers dead after projection GEMMs

    // 0) weight transpose+cast
    k_transpose_cast<<<(DIM_ * DIM_ + 255) / 256, 256, 0, stream>>>(w_in,  winT,  DIM_, DIM_);
    k_transpose_cast<<<(DIM_ * DIM_ + 255) / 256, 256, 0, stream>>>(w_out, woutT, DIM_, DIM_);

    // 1) layernorm + cast
    k_ln_cast<<<ROWS_ / 8, 256, 0, stream>>>(q, ln_w, ln_b, lnq, ROWS_, DIM_);
    k_ln_cast<<<ROWS_ / 8, 256, 0, stream>>>(k, ln_w, ln_b, lnk, ROWS_, DIM_);
    k_ln_cast<<<ROWS_ / 8, 256, 0, stream>>>(v, ln_w, ln_b, lnv, ROWS_, DIM_);

    // 2) projection GEMMs (bf16 out)
    dim3 gg(DIM_ / 128, ROWS_ / 128);
    k_gemm_bf16<false, false><<<gg, 256, 0, stream>>>(lnq, winT, nullptr, fqb, ROWS_, DIM_, DIM_);
    k_gemm_bf16<false, false><<<gg, 256, 0, stream>>>(lnk, winT, nullptr, fkb, ROWS_, DIM_, DIM_);
    k_gemm_bf16<false, false><<<gg, 256, 0, stream>>>(lnv, winT, nullptr, fvb, ROWS_, DIM_, DIM_);

    // 3) stats + pooled means
    k_zero_f32<<<(H_ * B_ * 2 * DH_ + 255) / 256, 256, 0, stream>>>(pool, H_ * B_ * 2 * DH_);
    dim3 gs(H_, B_ * N_);
    k_stats<<<gs, 32, 0, stream>>>(fqb, qm, iqn, pool, 0);
    k_stats<<<gs, 32, 0, stream>>>(fkb, km, ikn, pool, DH_);

    // 4) gate MLP
    k_gate<<<H_ * B_, DH_, 0, stream>>>(pool, wp_w1, wp_b1, wp_lw, wp_lb, wp_w2, wp_b2, wgate);

    // 5) fused attention
    dim3 ga(N_ / 128, H_ * B_);
    k_attn<<<ga, 256, 0, stream>>>(fqb, fkb, fvb, qm, iqn, km, ikn, wgate, Obuf);

    // 6) output GEMM (+bias, fp32 out)
    k_gemm_bf16<true, true><<<gg, 256, 0, stream>>>(Obuf, woutT, b_out, d_out, ROWS_, DIM_, DIM_);
}
